// AbstractDeformableConvolutional2D_52347061403996
// MI455X (gfx1250) — compile-verified, matrix operand order checked by reading, not run on hardware
//
#include <hip/hip_runtime.h>
#include <hip/hip_bf16.h>

// ---- CDNA5 (gfx1250) vector types for WMMA ----
typedef __attribute__((ext_vector_type(16))) __bf16 v16bf;
typedef __attribute__((ext_vector_type(8)))  __bf16 v8bf;
typedef __attribute__((ext_vector_type(2)))  __bf16 v2bf;
typedef __attribute__((ext_vector_type(8)))  float  v8f;

#define B_   8
#define H_   128
#define W_   128
#define C_   64
#define F_   64
#define NTAP 9
#define K_   (NTAP * C_)   // 576

// Padded LDS row strides (in elements) for conflict-free ds_load_b128:
// byte stride / 4 mod 64 == 36 banks -> 16 distinct bank-quads across 16 rows.
#define WSTRIDE 584        // >= 576, 1168 B rows, 16B-aligned fragments
#define ASTRIDE 72         // >= 64,   144 B rows, 16B-aligned fragments

// One block = 256 threads = 8 wave32; each wave computes a 16-pixel x 64-output
// tile of the implicit (131072 x 576) x (576 x 64) GEMM using
// v_wmma_f32_16x16x32_bf16. All 9 taps' weights are staged to LDS once
// (bf16, transposed [f][k]); after the single barrier the waves run fully
// decoupled (sA is wave-private; in-wave LDS ordering via DScnt suffices).
__global__ __launch_bounds__(256) void deform_conv2d_wmma_bf16(
    const float* __restrict__ xin,    // (B,H,W,C) f32
    const float* __restrict__ xoff,   // (B,H,W,9) f32  (y_offset is dead per reference)
    const float* __restrict__ wgt,    // (3,3,C,F) f32
    float* __restrict__ out)          // (B,H,W,F) f32
{
    __shared__ __align__(16) __bf16 sW[F_ * WSTRIDE];      // [f][k], all taps (~73 KB)
    __shared__ __align__(16) __bf16 sA[8][16 * ASTRIDE];   // per-wave samples [m][c] (~18 KB)

    const int t    = threadIdx.x;
    const int wave = t >> 5;
    const int lane = t & 31;
    const int Mr   = lane & 15;   // row/col within 16 (WMMA lane layout)
    const int half = lane >> 4;   // half-wave selects K+8 / M+8 groups

    const int pixBase = blockIdx.x * 128 + wave * 16;
    const int c0 = lane * 2;      // channel pair handled by this lane in sampling

    // ---- Stage ALL tap weights into LDS as bf16, transposed to [f][k] ----
    // wgt flat index i = k*F + f with k = n*C + c  -> sW[f*WSTRIDE + k]
    for (int i = t; i < K_ * F_; i += 256) {
        const int k = i >> 6;          // 0..575
        const int f = i & 63;
        sW[f * WSTRIDE + k] = (__bf16)wgt[i];   // coalesced global reads
    }
    __syncthreads();   // only barrier: sW is read-only afterwards

    v8f acc[4] = {v8f{}, v8f{}, v8f{}, v8f{}};

    for (int n = 0; n < NTAP; ++n) {
        // ---- Bilinear sampling: 16 pixels x 64 channels for this wave ----
        const float Rn = (float)((n % 3) - 1);   // R = [-1,0,1] x3 (faithful)
        for (int m = 0; m < 16; ++m) {
            const int p   = pixBase + m;
            const int b   = p >> 14;             // / (H*W)
            const int rem = p & 16383;
            const int h   = rem >> 7;
            const int w   = rem & 127;

            const float off = xoff[p * NTAP + n];
            const float xr = (float)(w + 1) + off + Rn;
            const float yr = (float)(h + 1) + off + Rn;   // uses x_offset: source bug

            int x0 = (int)floorf(xr), y0 = (int)floorf(yr);
            int x1 = x0 + 1,          y1 = y0 + 1;
            x0 = min(max(x0, 0), W_ + 1); x1 = min(max(x1, 0), W_ + 1);
            y0 = min(max(y0, 0), H_ + 1); y1 = min(max(y1, 0), H_ + 1);

            const float x0f = (float)x0, x1f = (float)x1;
            const float y0f = (float)y0, y1f = (float)y1;
            const float w0 = (y1f - yr) * (x1f - xr);
            const float w1 = (y1f - yr) * (xr - x0f);
            const float w2 = (yr - y0f) * (x1f - xr);
            const float w3 = (yr - y0f) * (xr - x0f);

            auto sample = [&](int yy, int xx) -> float2 {
                if (yy >= 1 && yy <= H_ && xx >= 1 && xx <= W_) {
                    const int idx = (((b * H_) + (yy - 1)) * W_ + (xx - 1)) * C_ + c0;
                    return *(const float2*)&xin[idx];   // coalesced b64 per lane
                }
                return make_float2(0.0f, 0.0f);         // zero padding
            };

            const float2 q0 = sample(y0, x0);
            const float2 q1 = sample(y0, x1);
            const float2 q2 = sample(y1, x0);
            const float2 q3 = sample(y1, x1);

            const float s0 = w0 * q0.x + w1 * q1.x + w2 * q2.x + w3 * q3.x;
            const float s1 = w0 * q0.y + w1 * q1.y + w2 * q2.y + w3 * q3.y;

            v2bf pk;
            pk[0] = (__bf16)s0;
            pk[1] = (__bf16)s1;
            *(v2bf*)&sA[wave][m * ASTRIDE + c0] = pk;   // packed b32 LDS store
        }

        // ---- WMMA: K = 64 channels of this tap = 2 chunks of 32 ----
        // (sA wave-private: DS in-order per wave, no barrier needed)
        #pragma unroll
        for (int kc = 0; kc < 2; ++kc) {
            // A fragment (16-bit A layout): lane Mr holds row M=Mr,
            // dwords 0-3 = K pairs at kbase+half*8, dwords 4-7 at +16.
            const __bf16* ap = &sA[wave][Mr * ASTRIDE + kc * 32 + half * 8];
            v8bf alo = *(const v8bf*)ap;          // ds_load_b128, conflict-free
            v8bf ahi = *(const v8bf*)(ap + 16);
            v16bf a;
            #pragma unroll
            for (int i = 0; i < 8; ++i) { a[i] = alo[i]; a[i + 8] = ahi[i]; }

            #pragma unroll
            for (int nt = 0; nt < 4; ++nt) {
                // B fragment mirrors A: column f = nt*16 + Mr over sW[f][k]
                const __bf16* bp =
                    &sW[(nt * 16 + Mr) * WSTRIDE + n * C_ + kc * 32 + half * 8];
                v8bf blo = *(const v8bf*)bp;
                v8bf bhi = *(const v8bf*)(bp + 16);
                v16bf bm;
                #pragma unroll
                for (int i = 0; i < 8; ++i) { bm[i] = blo[i]; bm[i + 8] = bhi[i]; }

                // D = A x B + C, f32 accumulate
                acc[nt] = __builtin_amdgcn_wmma_f32_16x16x32_bf16(
                    /*neg_a=*/false, a, /*neg_b=*/false, bm,
                    /*c_mod=*/(short)0, acc[nt],
                    /*reuse_a=*/false, /*reuse_b=*/false);
            }
        }
    }

    // ---- Store C/D: VGPR r -> row r + 8*half, lane%16 -> column ----
    #pragma unroll
    for (int nt = 0; nt < 4; ++nt) {
        #pragma unroll
        for (int r = 0; r < 8; ++r) {
            const int m = r + half * 8;
            const int f = nt * 16 + Mr;
            out[(size_t)(pixBase + m) * F_ + f] = acc[nt][r];
        }
    }
}

extern "C" void kernel_launch(void* const* d_in, const int* in_sizes, int n_in,
                              void* d_out, int out_size, void* d_ws, size_t ws_size,
                              hipStream_t stream) {
    (void)in_sizes; (void)n_in; (void)out_size; (void)d_ws; (void)ws_size;
    const float* x_in     = (const float*)d_in[0];
    const float* y_offset = (const float*)d_in[1];  // dead input (reference bug uses x_offset twice)
    const float* x_offset = (const float*)d_in[2];
    const float* kern     = (const float*)d_in[3];
    (void)y_offset;
    float* out = (float*)d_out;

    // 131072 pixels / (8 waves * 16 pixels) = 1024 blocks
    dim3 grid(1024), block(256);
    deform_conv2d_wmma_bf16<<<grid, block, 0, stream>>>(x_in, x_offset, kern, out);
}
